// SelfAttention_48292612276561
// MI455X (gfx1250) — matrix operand order repaired and checked by reading, not compiled, not run on hardware
//
#include <hip/hip_runtime.h>
#include <hip/hip_bf16.h>

// ---------------- problem constants ----------------
#define BATCH   16
#define CURRD   256
#define DIM     64
#define NPIX    16384        // 128*128
#define TN      128          // pixels per tile
#define NTILES  (NPIX / TN)  // 128

typedef __bf16 bf16;
typedef __attribute__((ext_vector_type(16))) __bf16 bf16x16;
typedef __attribute__((ext_vector_type(8)))  float  f32x8;
typedef __attribute__((ext_vector_type(4)))  unsigned int u32x4;
typedef __attribute__((ext_vector_type(8)))  int i32x8;
typedef __attribute__((ext_vector_type(4)))  int i32x4;
typedef __attribute__((address_space(3)))    bf16 lds_bf16;

#if __has_builtin(__builtin_amdgcn_tensor_load_to_lds) && \
    __has_builtin(__builtin_amdgcn_s_wait_tensorcnt)
#define USE_TDM 1
#endif

union FragU { bf16x16 v; uint4 q[2]; };

// ---- wave32 bf16 WMMA fragment loaders (ISA 7.12.2 layouts) ----
// A matrix 16x32: lane<16 -> row, K=8h..8h+7 and 16+8h..16+8h+7 (h = lane>>4)
__device__ __forceinline__ bf16x16 load_frag_a(const bf16* src, int stride,
                                               int m0, int k0, int lane) {
    int row = m0 + (lane & 15);
    int h8  = (lane >> 4) * 8;
    const bf16* base = src + row * stride + k0 + h8;
    FragU f;
    f.q[0] = *(const uint4*)(base);
    f.q[1] = *(const uint4*)(base + 16);
    return f.v;
}
// B matrix 32x16 read from a [N][K] (transposed) LDS tile:
// lane<16 -> col, K=8h..8h+7 and 16+8h..16+8h+7 (h = lane>>4)
__device__ __forceinline__ bf16x16 load_frag_b(const bf16* src, int stride,
                                               int n0, int k0, int lane) {
    int col = n0 + (lane & 15);
    int h16 = (lane >> 4) * 8;
    const bf16* base = src + col * stride + k0 + h16;
    FragU f;
    f.q[0] = *(const uint4*)(base);
    f.q[1] = *(const uint4*)(base + 16);
    return f.v;
}

__device__ __forceinline__ f32x8 wmma_bf16(bf16x16 a, bf16x16 b, f32x8 c) {
    return __builtin_amdgcn_wmma_f32_16x16x32_bf16(
        /*neg_a=*/false, a, /*neg_b=*/false, b,
        /*c_mod=*/(short)0, c, /*reuse_a=*/false, /*reuse_b=*/false);
}

// LDS strides (bf16 elements), padded so 16B vector loads stay aligned
#define S72   72    // 64  + 8
#define S136  136   // 128 + 8

// =====================================================================
// Kernel A: P = w_in@x + b_in  (store bf16 to ws);  Q,K;  partial score
// grid (NTILES, BATCH), 256 threads (8 waves)
// =====================================================================
__global__ __launch_bounds__(256) void qk_score_kernel(
    const float* __restrict__ x,      // [B][256][16384]
    const float* __restrict__ w_in,   // [64][256]
    const float* __restrict__ b_in,   // [64]
    const float* __restrict__ wq, const float* __restrict__ bq,
    const float* __restrict__ wk, const float* __restrict__ bk,
    bf16*  __restrict__ wsP,          // [B][NTILES][128][64] bf16
    float* __restrict__ score)        // [B][64][64] f32 (pre-zeroed)
{
    extern __shared__ __align__(16) char smem_raw[];
    bf16* Wc = (bf16*)smem_raw;        // [64][S72]   w_in chunk, A layout
    bf16* Xc = Wc + 64 * S72;          // [128][S72]  x chunk, [p][c] (B^T)
    bf16* Pl = Xc + 128 * S72;         // [128][S72]  P, [p][o] (B^T for GEMM2)
    bf16* Ql = Pl + 128 * S72;         // [64][S136]  Q, [i][p] (A for GEMM3)
    bf16* Kl = Ql + 64 * S136;         // [64][S136]  K, [j][p] (B^T for GEMM3)
    bf16* WQ = Kl + 64 * S136;         // [64][S72]
    bf16* WK = WQ + 64 * S72;          // [64][S72]
    float* biasS = (float*)(WK + 64 * S72);  // [192]: b_in | bq | bk

    const int tile = blockIdx.x, b = blockIdx.y;
    const int tid = threadIdx.x, lane = tid & 31, wave = tid >> 5;
    const int h = lane >> 4, n = lane & 15;
    const size_t pbase = (size_t)tile * TN;

    // stage wq / wk (fp32 -> bf16) and biases
    for (int i = tid; i < DIM * DIM; i += 256) {
        int r = i >> 6, c = i & 63;
        WQ[r * S72 + c] = (bf16)wq[i];
        WK[r * S72 + c] = (bf16)wk[i];
    }
    if (tid < DIM) {
        biasS[tid]           = b_in[tid];
        biasS[DIM + tid]     = bq[tid];
        biasS[2 * DIM + tid] = bk[tid];
    }

    // ---- GEMM1: P[64 x 128] = w_in @ x_tile, K=256 in 4 chunks of 64 ----
    const int ob = wave & 3;              // o-block 0..3
    const int pg = (wave >> 2) * 4;       // p-block group: 0 or 4
    f32x8 accP[4] = {};
    for (int ch = 0; ch < 4; ++ch) {
        const int c0 = ch * 64;
        __syncthreads();
        for (int i = tid; i < 64 * 64; i += 256) {           // Wc[o][c]
            int o = i >> 6, c = i & 63;
            Wc[o * S72 + c] = (bf16)w_in[o * CURRD + c0 + c];
        }
        for (int i = tid; i < 64 * TN; i += 256) {           // Xc[p][c]
            int c = i >> 7, p = i & (TN - 1);
            Xc[p * S72 + c] =
                (bf16)x[(size_t)(b * CURRD + c0 + c) * NPIX + pbase + p];
        }
        if (ch < 3) {
            // prefetch next chunk: 256 threads cover 64 rows x 4 cachelines
            const float* nx = &x[(size_t)(b * CURRD + c0 + 64 + (tid >> 2)) * NPIX
                                 + pbase + (tid & 3) * 32];
            __builtin_prefetch(nx, 0, 1);
        }
        __syncthreads();
        #pragma unroll
        for (int ks = 0; ks < 2; ++ks) {
            bf16x16 a = load_frag_a(Wc, S72, ob * 16, ks * 32, lane);
            #pragma unroll
            for (int q = 0; q < 4; ++q) {
                bf16x16 bb = load_frag_b(Xc, S72, (pg + q) * 16, ks * 32, lane);
                accP[q] = wmma_bf16(a, bb, accP[q]);
            }
        }
    }
    __syncthreads();
    // bias + store P to LDS [p][o]
    #pragma unroll
    for (int q = 0; q < 4; ++q) {
        int pt = (pg + q) * 16 + n;
        #pragma unroll
        for (int r = 0; r < 8; ++r) {
            int o = ob * 16 + r + 8 * h;
            Pl[pt * S72 + o] = (bf16)(accP[q][r] + biasS[o]);
        }
    }
    __syncthreads();
    // coalesced copy of P tile to workspace
    {
        bf16* dst = wsP + (size_t)(b * NTILES + tile) * TN * DIM;
        for (int i = tid; i < TN * DIM; i += 256)
            dst[i] = Pl[(i >> 6) * S72 + (i & 63)];
    }

    // ---- GEMM2: Q = wq@P + bq (waves 0-3), K = wk@P + bk (waves 4-7) ----
    {
        const bf16*  Wm   = (wave < 4) ? WQ : WK;
        const float* bias = biasS + ((wave < 4) ? DIM : 2 * DIM);
        bf16*        Out  = (wave < 4) ? Ql : Kl;
        const int    ob2  = wave & 3;
        for (int pb = 0; pb < 8; ++pb) {
            f32x8 acc = {};
            #pragma unroll
            for (int ks = 0; ks < 2; ++ks) {
                bf16x16 a  = load_frag_a(Wm, S72, ob2 * 16, ks * 32, lane);
                bf16x16 bb = load_frag_b(Pl, S72, pb * 16, ks * 32, lane);
                acc = wmma_bf16(a, bb, acc);
            }
            int pt = pb * 16 + n;
            #pragma unroll
            for (int r = 0; r < 8; ++r) {
                int ii = ob2 * 16 + r + 8 * h;
                Out[ii * S136 + pt] = (bf16)(acc[r] + bias[ii]);
            }
        }
    }
    __syncthreads();

    // ---- GEMM3: partial S[64x64] = Q (tile) . K^T, atomically reduced ----
    float* sc = score + (size_t)b * DIM * DIM;
    #pragma unroll
    for (int t = 0; t < 2; ++t) {
        int tt = wave * 2 + t;
        int ib = tt >> 2, jb = tt & 3;
        f32x8 acc = {};
        #pragma unroll
        for (int ks = 0; ks < 4; ++ks) {
            bf16x16 a  = load_frag_a(Ql, S136, ib * 16, ks * 32, lane);
            bf16x16 bb = load_frag_b(Kl, S136, jb * 16, ks * 32, lane);
            acc = wmma_bf16(a, bb, acc);
        }
        #pragma unroll
        for (int r = 0; r < 8; ++r) {
            int ii = ib * 16 + r + 8 * h;
            unsafeAtomicAdd(&sc[ii * DIM + jb * 16 + n], acc[r]);
        }
    }
}

// =====================================================================
// Kernel B: row softmax of scale*score, write attn^T (bf16)
// =====================================================================
__global__ void softmax_kernel(const float* __restrict__ score,
                               bf16* __restrict__ attnT) {
    const int b = blockIdx.x, i = threadIdx.x;
    const float scale = 0.125f;               // 1/sqrt(64)
    const float* row = score + (size_t)b * DIM * DIM + i * DIM;
    float v[DIM], m = -INFINITY;
    #pragma unroll
    for (int j = 0; j < DIM; ++j) { v[j] = row[j] * scale; m = fmaxf(m, v[j]); }
    float s = 0.f;
    #pragma unroll
    for (int j = 0; j < DIM; ++j) { v[j] = __expf(v[j] - m); s += v[j]; }
    const float inv = 1.f / s;
    bf16* at = attnT + (size_t)b * DIM * DIM;
    #pragma unroll
    for (int j = 0; j < DIM; ++j) at[j * DIM + i] = (bf16)(v[j] * inv);
}

// =====================================================================
// Kernel C: V = wv@P + bv; Wt = attn^T @ V; out = w_out@Wt + b_out + x
// grid (NTILES, BATCH), 256 threads
// =====================================================================
__global__ __launch_bounds__(256) void out_kernel(
    const float* __restrict__ x,
    const bf16*  __restrict__ wsP,
    const float* __restrict__ wv, const float* __restrict__ bv,
    const bf16*  __restrict__ attnT,
    const float* __restrict__ w_out, const float* __restrict__ b_out,
    float* __restrict__ out)
{
    extern __shared__ __align__(16) char smem_raw[];
    bf16* Pl = (bf16*)smem_raw;        // [128][S72] [p][o]
    bf16* WV = Pl + 128 * S72;         // [64][S72]
    bf16* Vl = WV + 64 * S72;          // [128][S72] [p][c]
    bf16* AT = Vl + 128 * S72;         // [64][S72]  attn^T row-major [j][c]
    bf16* Wt = AT + 64 * S72;          // [128][S72] [p][j]
    bf16* WO = Wt + 128 * S72;         // [256][S72]
    float* biasS = (float*)(WO + 256 * S72); // [320]: bv | b_out

    const int tile = blockIdx.x, b = blockIdx.y;
    const int tid = threadIdx.x, lane = tid & 31, wave = tid >> 5;
    const int h = lane >> 4, n = lane & 15;
    const size_t pbase = (size_t)tile * TN;

    const bf16* src = wsP + (size_t)(b * NTILES + tile) * TN * DIM;
#if defined(USE_TDM)
    // --- Tensor Data Mover: DMA the 128x64 bf16 P tile into LDS, padding
    // each 64-elem (128B = 32-DWORD) row by 8 elems (4 DWORDs) -> stride S72.
    if (wave == 0) {
        unsigned      ldsOff = (unsigned)(uintptr_t)(lds_bf16*)Pl;
        unsigned long long ga = (unsigned long long)(uintptr_t)src;
        u32x4 g0;
        g0[0] = 1u;                                   // count=1, user mode
        g0[1] = ldsOff;                               // lds_addr (bytes)
        g0[2] = (unsigned)(ga & 0xffffffffu);         // global_addr[31:0]
        g0[3] = (unsigned)((ga >> 32) & 0x1ffffffu)   // global_addr[56:32]
                | (2u << 30);                         // type = 2 (image)
        i32x8 g1 = {};
        g1[0] = (1 << 16)        // data_size = 1 (2 bytes)
              | (1 << 20)        // pad_enable
              | (4 << 22)        // pad_interval: 32 DWORDs
              | (3 << 25);       // pad_amount:   4 DWORDs
        g1[1] = (DIM << 16);     // tensor_dim0 = 64 (low 16 bits)
        g1[2] = (TN << 16);      // tensor_dim1 = 128 (low 16 bits)
        g1[3] = (DIM << 16);     // tile_dim0 = 64
        g1[4] = TN;              // tile_dim1 = 128, tile_dim2 = 0
        g1[5] = DIM;             // tensor_dim0_stride = 64 (low 32)
        g1[6] = 0;               // stride hi, tensor_dim1_stride = 0
        g1[7] = 0;
        i32x4 gz4 = {};
        i32x8 gz8 = {};
        // 6-arg form (clang-23 / therock-10.0 headers): groups 0..3 + extra + cpol
        __builtin_amdgcn_tensor_load_to_lds(g0, g1, gz4, gz4, gz8, 0);
        __builtin_amdgcn_s_wait_tensorcnt(0);
    }
#else
    for (int i = tid; i < TN * DIM; i += 256)
        Pl[(i >> 6) * S72 + (i & 63)] = src[i];
#endif
    for (int i = tid; i < DIM * DIM; i += 256) {
        WV[(i >> 6) * S72 + (i & 63)] = (bf16)wv[i];
        AT[(i >> 6) * S72 + (i & 63)] = attnT[(size_t)b * DIM * DIM + i];
    }
    for (int i = tid; i < CURRD * DIM; i += 256)
        WO[(i >> 6) * S72 + (i & 63)] = (bf16)w_out[i];
    if (tid < DIM) biasS[tid] = bv[tid];
    if (tid < CURRD) biasS[DIM + tid] = b_out[tid];
    __syncthreads();

    const int ob = wave & 3, pg = (wave >> 2) * 4;
    // ---- V = wv @ P + bv ----
    #pragma unroll
    for (int q = 0; q < 4; ++q) {
        int pb = pg + q;
        f32x8 acc = {};
        #pragma unroll
        for (int ks = 0; ks < 2; ++ks) {
            bf16x16 a  = load_frag_a(WV, S72, ob * 16, ks * 32, lane);
            bf16x16 bb = load_frag_b(Pl, S72, pb * 16, ks * 32, lane);
            acc = wmma_bf16(a, bb, acc);
        }
        int pt = pb * 16 + n;
        #pragma unroll
        for (int r = 0; r < 8; ++r) {
            int c = ob * 16 + r + 8 * h;
            Vl[pt * S72 + c] = (bf16)(acc[r] + biasS[c]);
        }
    }
    __syncthreads();
    // ---- Wt = attn^T @ V ----
    #pragma unroll
    for (int q = 0; q < 4; ++q) {
        int pb = pg + q;
        f32x8 acc = {};
        #pragma unroll
        for (int ks = 0; ks < 2; ++ks) {
            bf16x16 a  = load_frag_a(AT, S72, ob * 16, ks * 32, lane);
            bf16x16 bb = load_frag_b(Vl, S72, pb * 16, ks * 32, lane);
            acc = wmma_bf16(a, bb, acc);
        }
        int pt = pb * 16 + n;
        #pragma unroll
        for (int r = 0; r < 8; ++r) {
            int j = ob * 16 + r + 8 * h;
            Wt[pt * S72 + j] = (bf16)acc[r];
        }
    }
    __syncthreads();
    // ---- out = w_out @ Wt + b_out + x (residual) ----
    for (int t = 0; t < 16; ++t) {
        int ob2 = 2 * wave + (t >> 3);
        int pb  = t & 7;
        f32x8 acc = {};
        #pragma unroll
        for (int ks = 0; ks < 2; ++ks) {
            bf16x16 a  = load_frag_a(WO, S72, ob2 * 16, ks * 32, lane);
            bf16x16 bb = load_frag_b(Wt, S72, pb * 16, ks * 32, lane);
            acc = wmma_bf16(a, bb, acc);
        }
        int pt = pb * 16 + n;
        #pragma unroll
        for (int r = 0; r < 8; ++r) {
            int o = ob2 * 16 + r + 8 * h;
            size_t idx = (size_t)(b * CURRD + o) * NPIX + pbase + pt;
            out[idx] = acc[r] + biasS[DIM + o] + x[idx];
        }
    }
}

// =====================================================================
extern "C" void kernel_launch(void* const* d_in, const int* in_sizes, int n_in,
                              void* d_out, int out_size, void* d_ws, size_t ws_size,
                              hipStream_t stream) {
    const float* x     = (const float*)d_in[0];
    const float* w_in  = (const float*)d_in[1];
    const float* b_in  = (const float*)d_in[2];
    const float* wq    = (const float*)d_in[3];
    const float* bq    = (const float*)d_in[4];
    const float* wk    = (const float*)d_in[5];
    const float* bk    = (const float*)d_in[6];
    const float* wv    = (const float*)d_in[7];
    const float* bv    = (const float*)d_in[8];
    const float* w_out = (const float*)d_in[9];
    const float* b_out = (const float*)d_in[10];
    float* out = (float*)d_out;

    // workspace: P (bf16, 32MB) | score (f32) | attnT (bf16)
    const size_t pBytes = (size_t)BATCH * NPIX * DIM * sizeof(bf16);
    bf16*  wsP   = (bf16*)d_ws;
    float* score = (float*)((char*)d_ws + pBytes);
    bf16*  attnT = (bf16*)((char*)score + (size_t)BATCH * DIM * DIM * sizeof(float));

    (void)hipMemsetAsync(score, 0, (size_t)BATCH * DIM * DIM * sizeof(float), stream);

    const size_t ldsA = (size_t)(64*S72 + 128*S72 + 128*S72 + 64*S136 + 64*S136
                                 + 64*S72 + 64*S72) * sizeof(bf16)
                        + 192 * sizeof(float);                       // ~100 KB
    const size_t ldsC = (size_t)(128*S72*3 + 64*S72*2 + 256*S72) * sizeof(bf16)
                        + 320 * sizeof(float);                       // ~112 KB

    dim3 grid(NTILES, BATCH);
    qk_score_kernel<<<grid, 256, ldsA, stream>>>(x, w_in, b_in, wq, bq, wk, bk,
                                                 wsP, score);
    softmax_kernel<<<BATCH, 64, 0, stream>>>(score, attnT);
    out_kernel<<<grid, 256, ldsC, stream>>>(x, wsP, wv, bv, attnT, w_out, b_out,
                                            out);
}
